// Traj_Model_Incremental_52561809768849
// MI455X (gfx1250) — matrix operand order repaired and checked by reading, not compiled, not run on hardware
//
#include <hip/hip_runtime.h>
#include <hip/hip_bf16.h>

typedef __attribute__((ext_vector_type(16))) __bf16 v16bf;
typedef __attribute__((ext_vector_type(2)))  __bf16 v2bf;
typedef __attribute__((ext_vector_type(8)))  float  v8f;

#define NTOK   8192          // B*T
#define DMODEL 768
#define DFF    3072
#define NEXP   10
#define LINSZ  1376
#define MAXMT  256           // max 64-row tiles per expert (16384/64)
#define APITCH 40            // A LDS pitch: 80B rows -> b128-aligned, 20m%64 distinct
#define BPITCH 34            // B LDS pitch: 17n%64 distinct for fragment reads

// ---------------------------------------------------------------- utils
__device__ __forceinline__ unsigned short bfbits(float f) {
  __bf16 b = (__bf16)f;
  return __builtin_bit_cast(unsigned short, b);
}
__device__ __forceinline__ __bf16 bits2bf(unsigned short u) {
  return __builtin_bit_cast(__bf16, u);
}
__device__ __forceinline__ unsigned packbf(float a, float b) {
  v2bf p; p.x = (__bf16)a; p.y = (__bf16)b;      // -> v_cvt_pk_bf16_f32
  return __builtin_bit_cast(unsigned, p);
}

// ---------------------------------------------------------------- init
__global__ void init_counts_kernel(int* counts) {
  if (threadIdx.x < NEXP) counts[threadIdx.x] = 0;
}

// ---------------------------------------------------------------- fp32 -> bf16 weights
__global__ void convert_weights_kernel(const float* __restrict__ w1,
                                       const float* __restrict__ w2,
                                       unsigned short* __restrict__ w1b,
                                       unsigned short* __restrict__ w2b) {
  const long long W = (long long)NEXP * DMODEL * DFF;
  long long i = (long long)blockIdx.x * blockDim.x + threadIdx.x;
  const long long stride = (long long)gridDim.x * blockDim.x;
  for (; i < 2 * W; i += stride) {
    if (i < W) w1b[i]     = bfbits(w1[i]);
    else       w2b[i - W] = bfbits(w2[i - W]);
  }
}

// ---------------------------------------------------------------- router
// one wave32 per token
__global__ void router_kernel(const float* __restrict__ x, const int* __restrict__ city_p,
                              const float* __restrict__ dt, const float* __restrict__ dd,
                              const float* __restrict__ drg, const float* __restrict__ dent,
                              const float* __restrict__ city_emb,
                              const float* __restrict__ rw, const float* __restrict__ rb,
                              float* __restrict__ out, float* __restrict__ gate1_out,
                              int* __restrict__ counts,
                              int* __restrict__ top2e, float* __restrict__ top2g) {
  const int wave = threadIdx.x >> 5;
  const int lane = threadIdx.x & 31;
  const int tok  = blockIdx.x * 8 + wave;
  if (tok >= NTOK) return;
  const int city = *city_p;

  float acc[NEXP];
#pragma unroll
  for (int e = 0; e < NEXP; e++) acc[e] = 0.f;

  for (int f = lane; f < LINSZ; f += 32) {
    float v;
    if      (f <  768) v = x[tok * DMODEL + f];
    else if (f <  800) v = city_emb[city * 32 + (f - 768)];
    else if (f <  992) v = dt[tok * 192 + (f - 800)];
    else if (f < 1184) v = dd[tok * 192 + (f - 992)];
    else if (f < 1280) v = drg[tok * 96 + (f - 1184)];
    else               v = dent[tok * 96 + (f - 1280)];
#pragma unroll
    for (int e = 0; e < NEXP; e++) acc[e] = fmaf(v, rw[e * LINSZ + f], acc[e]);
  }
#pragma unroll
  for (int off = 16; off > 0; off >>= 1)
#pragma unroll
    for (int e = 0; e < NEXP; e++) acc[e] += __shfl_xor(acc[e], off, 32);
#pragma unroll
  for (int e = 0; e < NEXP; e++) acc[e] += rb[e];

  // full softmax -> gate1
  float mx = acc[0];
#pragma unroll
  for (int e = 1; e < NEXP; e++) mx = fmaxf(mx, acc[e]);
  float ex[NEXP]; float s = 0.f;
#pragma unroll
  for (int e = 0; e < NEXP; e++) { ex[e] = expf(acc[e] - mx); s += ex[e]; }
  const float inv = 1.f / s;
#pragma unroll
  for (int q = 0; q < NEXP; q++)
    if (lane == q) gate1_out[tok * NEXP + q] = ex[q] * inv;

  // top-2 (lowest index wins ties, like jax.lax.top_k)
  int e1 = 0; float v1 = acc[0];
#pragma unroll
  for (int e = 1; e < NEXP; e++) if (acc[e] > v1) { v1 = acc[e]; e1 = e; }
  int e2 = -1; float v2 = -3.4e38f;
#pragma unroll
  for (int e = 0; e < NEXP; e++)
    if (e != e1 && acc[e] > v2) { v2 = acc[e]; e2 = e; }
  const float t  = expf(v2 - v1);          // v1 >= v2
  const float g1 = 1.f / (1.f + t);
  const float g2 = t / (1.f + t);

  if (lane == 0) {
    top2e[tok * 2 + 0] = e1; top2g[tok * 2 + 0] = g1;
    top2e[tok * 2 + 1] = e2; top2g[tok * 2 + 1] = g2;
    atomicAdd(&counts[e1], 1);
    atomicAdd(&counts[e2], 1);
  }
  // zero our output row (accumulated later via atomics)
  for (int c = lane; c < DMODEL; c += 32) out[(long long)tok * DMODEL + c] = 0.f;
}

// ---------------------------------------------------------------- scan + scatter
__global__ void scan_kernel(const int* __restrict__ counts, int* __restrict__ offs,
                            int* __restrict__ cursor) {
  if (threadIdx.x == 0 && blockIdx.x == 0) {
    int a = 0;
    for (int e = 0; e < NEXP; e++) { offs[e] = a; cursor[e] = a; a += counts[e]; }
    offs[NEXP] = a;
  }
}

__global__ void scatter_kernel(const int* __restrict__ top2e, const float* __restrict__ top2g,
                               int* __restrict__ cursor,
                               int* __restrict__ tok_of, float* __restrict__ gate_of) {
  const int tok = blockIdx.x * blockDim.x + threadIdx.x;
  if (tok >= NTOK) return;
#pragma unroll
  for (int k = 0; k < 2; k++) {
    const int e = top2e[tok * 2 + k];
    const int slot = atomicAdd(&cursor[e], 1);
    tok_of[slot]  = tok;
    gate_of[slot] = top2g[tok * 2 + k];
  }
}

// ---------------------------------------------------------------- gathered expert GEMM
// Block: 256 thr = 8 waves; tile 64(M) x 128(N); K staged 32/step, double-buffered LDS.
// Waves arranged 2(m) x 4(n); each wave owns a 32x32 quad = 4 wmma tiles (2 A-frags
// reused over 2 B-frags -> 1.0 LDS frag loads per v_wmma_f32_16x16x32_bf16).
// PASS2=false: H = gelu(X @ W1 + b1)    PASS2=true: out += gate * (H @ W2 + b2)
template <int KDIM, int NDIM, bool PASS2>
__global__ __launch_bounds__(256)
void moe_gemm_kernel(const unsigned short* __restrict__ Wb,
                     const float* __restrict__ bias,
                     const float* __restrict__ x,
                     const unsigned short* __restrict__ Hin,
                     unsigned short* __restrict__ Hout,
                     float* __restrict__ out,
                     const int* __restrict__ offs,
                     const int* __restrict__ counts,
                     const int* __restrict__ tok_of,
                     const float* __restrict__ gate_of) {
  const int e     = blockIdx.x >> 8;
  const int mt    = blockIdx.x & (MAXMT - 1);
  const int cnt   = counts[e];
  const int mbase = mt * 64;
  if (mbase >= cnt) return;
  const int slot0 = offs[e] + mbase;
  const int rmax  = cnt - mbase - 1;   // last valid local row (>= 0)
  const int nbase = blockIdx.y * 128;

  const int tid  = threadIdx.x;
  const int lane = tid & 31;
  const int wave = tid >> 5;
  const int wm   = wave & 1;           // m quad: rows [wm*32, wm*32+32)
  const int wn   = wave >> 1;          // n quad: cols [wn*32, wn*32+32)

  __shared__ __attribute__((aligned(16))) __bf16 Abuf[2][64][APITCH];
  __shared__ __attribute__((aligned(16))) __bf16 Bbuf[2][128][BPITCH];

  const __bf16* WbT = (const __bf16*)Wb;
  const __bf16* wbase = WbT + (long long)e * KDIM * NDIM + nbase;

  // ---- staging coordinates (all loads clamped -> branchless, unconditional)
  const int rB  = tid >> 4;           // B k-rows: rB and rB+16
  const int cB  = (tid & 15) * 8;     // B col 0,8,..,120 (16B chunk)
  // pass1 A: two float4 rows
  const int rA0 = tid >> 3;           // 0..31
  const int rA1 = rA0 + 32;           // 32..63
  const int cgA = (tid & 7) * 4;      // col 0,4,..,28
  // pass2 A: one uint4 row
  const int rA  = tid >> 2;           // 0..63
  const int gA  = (tid & 3) * 8;      // col 0,8,16,24

  long long aoff0 = 0, aoff1 = 0, aoff2 = 0;
  if (PASS2) {
    const int rc = (rA <= rmax) ? rA : rmax;
    aoff2 = (long long)(slot0 + rc) * DFF + gA;
  } else {
    const int rc0 = (rA0 <= rmax) ? rA0 : rmax;
    const int rc1 = (rA1 <= rmax) ? rA1 : rmax;
    aoff0 = (long long)tok_of[slot0 + rc0] * DMODEL + cgA;
    aoff1 = (long long)tok_of[slot0 + rc1] * DMODEL + cgA;
  }

  float4 fa0 = {}, fa1 = {};
  uint4  ua = {}, ub0 = {}, ub1 = {};

  auto gload = [&](int ks) {
    ub0 = *(const uint4*)(wbase + (long long)(ks * 32 + rB) * NDIM + cB);
    ub1 = *(const uint4*)(wbase + (long long)(ks * 32 + rB + 16) * NDIM + cB);
    if (PASS2) {
      ua = *(const uint4*)(Hin + aoff2 + ks * 32);
    } else {
      fa0 = *(const float4*)(x + aoff0 + ks * 32);
      fa1 = *(const float4*)(x + aoff1 + ks * 32);
    }
  };
  auto sstore = [&](int b) {
    // B transpose: Bbuf[n][k]
    const unsigned u0[4] = {ub0.x, ub0.y, ub0.z, ub0.w};
    const unsigned u1[4] = {ub1.x, ub1.y, ub1.z, ub1.w};
#pragma unroll
    for (int q = 0; q < 4; q++) {
      Bbuf[b][cB + 2 * q][rB]          = bits2bf((unsigned short)(u0[q] & 0xffffu));
      Bbuf[b][cB + 2 * q + 1][rB]      = bits2bf((unsigned short)(u0[q] >> 16));
      Bbuf[b][cB + 2 * q][rB + 16]     = bits2bf((unsigned short)(u1[q] & 0xffffu));
      Bbuf[b][cB + 2 * q + 1][rB + 16] = bits2bf((unsigned short)(u1[q] >> 16));
    }
    if (PASS2) {
      *(uint4*)&Abuf[b][rA][gA] = ua;                        // ds_store_b128
    } else {
      uint2 s0 = {packbf(fa0.x, fa0.y), packbf(fa0.z, fa0.w)};
      uint2 s1 = {packbf(fa1.x, fa1.y), packbf(fa1.z, fa1.w)};
      *(uint2*)&Abuf[b][rA0][cgA] = s0;                      // ds_store_b64
      *(uint2*)&Abuf[b][rA1][cgA] = s1;
    }
  };

  v8f acc00 = {}, acc01 = {}, acc10 = {}, acc11 = {};

  // fragment addressing (ISA 7.12.2, bf16, wave32)
  const int am0 = wm * 32 + (lane & 15);         // m-tile 0 row
  const int am1 = am0 + 16;                      // m-tile 1 row
  const int akb = (lane < 16) ? 0 : 8;
  const int bn0 = wn * 32 + (lane & 15);         // n-tile 0 col
  const int bkh = (lane < 16) ? 0 : 8;

  const int KT = KDIM / 32;
  gload(0);
  sstore(0);
  for (int ks = 0; ks < KT; ks++) {
    const int cur = ks & 1;
    __syncthreads();
    // issue next tile's global loads first (latency hidden by WMMA work)
    const int ksn = (ks + 1 < KT) ? ks + 1 : 0;
    gload(ksn);

    union { v16bf v; unsigned u[8]; } af0, af1, bf0, bf1;
#pragma unroll
    for (int q = 0; q < 8; q++) {
      const int k = akb + (q < 4 ? 2 * q : 16 + 2 * (q - 4));
      af0.u[q] = *(const unsigned*)&Abuf[cur][am0][k];
      af1.u[q] = *(const unsigned*)&Abuf[cur][am1][k];
    }
#pragma unroll
    for (int q = 0; q < 8; q++) {
      const int k = bkh + (q < 4 ? 2 * q : 16 + 2 * (q - 4));
      bf0.u[q] = *(const unsigned*)&Bbuf[cur][bn0][k];
      bf1.u[q] = *(const unsigned*)&Bbuf[cur][bn0 + 16][k];
    }
    acc00 = __builtin_amdgcn_wmma_f32_16x16x32_bf16(false, af0.v, false, bf0.v,
                                                    (short)0, acc00, false, false);
    acc01 = __builtin_amdgcn_wmma_f32_16x16x32_bf16(false, af0.v, false, bf1.v,
                                                    (short)0, acc01, false, false);
    acc10 = __builtin_amdgcn_wmma_f32_16x16x32_bf16(false, af1.v, false, bf0.v,
                                                    (short)0, acc10, false, false);
    acc11 = __builtin_amdgcn_wmma_f32_16x16x32_bf16(false, af1.v, false, bf1.v,
                                                    (short)0, acc11, false, false);
    // stage next tile into the other buffer
    sstore(cur ^ 1);
  }

  // epilogue: D layout — lanes<16: N=lane, M=vgpr; lanes>=16: N=lane-16, M=vgpr+8
  const int mh = (lane >> 4) * 8;
#pragma unroll
  for (int mi = 0; mi < 2; mi++) {
#pragma unroll
    for (int nt = 0; nt < 2; nt++) {
      v8f acc = mi ? (nt ? acc11 : acc10) : (nt ? acc01 : acc00);
      const int   gn = nbase + wn * 32 + nt * 16 + (lane & 15);
      const float bv = bias[e * NDIM + gn];
#pragma unroll
      for (int i = 0; i < 8; i++) {
        const int r = wm * 32 + mi * 16 + mh + i;
        if (r <= rmax) {
          float v = acc[i] + bv;
          if (PASS2) {
            const int slot = slot0 + r;
            unsafeAtomicAdd(&out[(long long)tok_of[slot] * DMODEL + gn], v * gate_of[slot]);
          } else {
            const float g = 0.5f * v * (1.f + erff(v * 0.70710678118654752440f));
            Hout[(long long)(slot0 + r) * DFF + gn] = bfbits(g);
          }
        }
      }
    }
  }
}

// ---------------------------------------------------------------- launcher
extern "C" void kernel_launch(void* const* d_in, const int* in_sizes, int n_in,
                              void* d_out, int out_size, void* d_ws, size_t ws_size,
                              hipStream_t stream) {
  (void)in_sizes; (void)n_in; (void)out_size; (void)ws_size;
  const float* x        = (const float*)d_in[0];
  const int*   city     = (const int*)d_in[1];
  const float* dt       = (const float*)d_in[2];
  const float* dd       = (const float*)d_in[3];
  const float* drg      = (const float*)d_in[4];
  const float* dent     = (const float*)d_in[5];
  const float* city_emb = (const float*)d_in[6];
  const float* rw       = (const float*)d_in[7];
  const float* rb       = (const float*)d_in[8];
  const float* w1       = (const float*)d_in[9];
  const float* b1       = (const float*)d_in[10];
  const float* w2       = (const float*)d_in[11];
  const float* b2       = (const float*)d_in[12];

  float* out   = (float*)d_out;                        // [8192, 768]
  float* gate1 = out + (size_t)NTOK * DMODEL;          // [8192, 10]

  // workspace carve-up (256B aligned)
  char* ws = (char*)d_ws;
  size_t off = 0;
  auto carve = [&](size_t bytes) {
    void* p = ws + off;
    off = (off + bytes + 255) & ~(size_t)255;
    return p;
  };
  const size_t W = (size_t)NEXP * DMODEL * DFF;                 // 23.6M elems
  unsigned short* w1b     = (unsigned short*)carve(W * 2);      // 47.2 MB
  unsigned short* w2b     = (unsigned short*)carve(W * 2);      // 47.2 MB
  unsigned short* H       = (unsigned short*)carve((size_t)2 * NTOK * DFF * 2); // 100.7 MB
  int*            counts  = (int*)carve(NEXP * 4);
  int*            offs    = (int*)carve((NEXP + 1) * 4);
  int*            cursor  = (int*)carve(NEXP * 4);
  int*            tok_of  = (int*)carve((size_t)2 * NTOK * 4);
  float*          gate_of = (float*)carve((size_t)2 * NTOK * 4);
  int*            top2e   = (int*)carve((size_t)2 * NTOK * 4);
  float*          top2g   = (float*)carve((size_t)2 * NTOK * 4);

  init_counts_kernel<<<1, 32, 0, stream>>>(counts);
  convert_weights_kernel<<<8192, 256, 0, stream>>>(w1, w2, w1b, w2b);
  router_kernel<<<NTOK / 8, 256, 0, stream>>>(x, city, dt, dd, drg, dent, city_emb, rw, rb,
                                              out, gate1, counts, top2e, top2g);
  scan_kernel<<<1, 32, 0, stream>>>(counts, offs, cursor);
  scatter_kernel<<<NTOK / 256, 256, 0, stream>>>(top2e, top2g, cursor, tok_of, gate_of);

  // pass 1: H = gelu(X @ W1 + b1)
  moe_gemm_kernel<DMODEL, DFF, false>
      <<<dim3(NEXP * MAXMT, DFF / 128), 256, 0, stream>>>(
          w1b, b1, x, nullptr, H, nullptr, offs, counts, tok_of, gate_of);
  // pass 2: out += gate * (H @ W2 + b2)
  moe_gemm_kernel<DFF, DMODEL, true>
      <<<dim3(NEXP * MAXMT, DMODEL / 128), 256, 0, stream>>>(
          w2b, b2, nullptr, H, nullptr, out, offs, counts, tok_of, gate_of);
}
